// MultiHeadAttention_25031069401563
// MI455X (gfx1250) — matrix operand order repaired and checked by reading, not compiled, not run on hardware
//
#include <hip/hip_runtime.h>
#include <hip/hip_bf16.h>
#include <math.h>

// ---------------------------------------------------------------------------
// MI455X (gfx1250) fused GQA attention block, bf16 WMMA (f32 accumulate).
// Pipeline: cvt(f32->bf16) -> QKV GEMMs -> RMSNorm+RoPE (+V transpose)
//           -> flash attention (async K/V staging to LDS) -> O-proj GEMM.
// ---------------------------------------------------------------------------

typedef __bf16 bf16_t;
typedef bf16_t v16bf __attribute__((ext_vector_type(16)));
typedef bf16_t v8bf  __attribute__((ext_vector_type(8)));
typedef float  v8f   __attribute__((ext_vector_type(8)));

// pointer types matching __builtin_amdgcn_global_load_async_to_lds_b128:
// (global v4i*, lds v4i*, imm offset, imm cpol)
typedef int v4i_vs __attribute__((__vector_size__(16)));
typedef __attribute__((address_space(1))) v4i_vs GlobalV4;
typedef __attribute__((address_space(3))) v4i_vs LocalV4;

#define S_LEN   2048
#define HDIM    2048
#define NHEAD   16
#define NKVH    4
#define HD      128
#define BATCH   2
#define MTOT    (BATCH * S_LEN)   // 4096 rows

// padded LDS row lengths (elements) to spread WMMA fragment rows across banks
#define KROW 136   // 32x128 K tile rows
#define VROW 40    // 128x32 V^T tile rows
#define PROW 40    // 16x32 P tile rows

#define HAS_ASYNC_LDS __has_builtin(__builtin_amdgcn_global_load_async_to_lds_b128)

// round-to-nearest-even f32 -> bf16 bit pattern
__device__ __forceinline__ unsigned short f2bf(float f) {
  unsigned u = __float_as_uint(f);
  u += 0x7FFFu + ((u >> 16) & 1u);
  return (unsigned short)(u >> 16);
}

// A-fragment (16x32 bf16): slots 0..15 = K(half*16 .. +15)
__device__ __forceinline__ v16bf make_afrag(const __hip_bfloat16* p, int half) {
  const v8bf lo = *reinterpret_cast<const v8bf*>(p + half * 16);
  const v8bf hi = *reinterpret_cast<const v8bf*>(p + half * 16 + 8);
  return __builtin_shufflevector(lo, hi, 0,1,2,3,4,5,6,7,8,9,10,11,12,13,14,15);
}
// B-fragment, 32-byte aligned source (global, unpadded strides)
__device__ __forceinline__ v16bf load_bfrag(const __hip_bfloat16* p) {
  return *reinterpret_cast<const v16bf*>(p);
}
// B-fragment from 16-byte aligned source (padded LDS rows): two b128 loads
__device__ __forceinline__ v16bf load_bfrag16(const __hip_bfloat16* p) {
  const v8bf lo = *reinterpret_cast<const v8bf*>(p);
  const v8bf hi = *reinterpret_cast<const v8bf*>(p + 8);
  return __builtin_shufflevector(lo, hi, 0,1,2,3,4,5,6,7,8,9,10,11,12,13,14,15);
}
__device__ __forceinline__ v8f wmma_bf16(v16bf a, v16bf b, v8f c) {
  return __builtin_amdgcn_wmma_f32_16x16x32_bf16(false, a, false, b, (short)0, c,
                                                 false, false);
}

// 16-byte global -> LDS copy: async LDS-DMA (ASYNCcnt-tracked) when the
// builtin is available, else load+ds_store fallback.
__device__ __forceinline__ void cp16_g2l(const __hip_bfloat16* g, __hip_bfloat16* l) {
#if HAS_ASYNC_LDS
  __builtin_amdgcn_global_load_async_to_lds_b128(
      (GlobalV4*)(unsigned long long)(uintptr_t)g,
      (LocalV4*)(unsigned int)(uintptr_t)l, 0, 0);
#else
  *reinterpret_cast<uint4*>(l) = *reinterpret_cast<const uint4*>(g);
#endif
}
__device__ __forceinline__ void wait_async_lds() {
#if HAS_ASYNC_LDS
  __asm__ volatile("s_wait_asynccnt 0" ::: "memory");
#endif
}

// ---------------------------------------------------------------------------
// Kernel 0: fp32 -> bf16 conversion (4 elems / thread, packed 8-byte stores)
// ---------------------------------------------------------------------------
__global__ __launch_bounds__(256) void cvt_f32_to_bf16(const float* __restrict__ src,
                                                       uint2* __restrict__ dst, int n4) {
  int i = blockIdx.x * blockDim.x + threadIdx.x;
  if (i >= n4) return;
  const float4 v = reinterpret_cast<const float4*>(src)[i];
  uint2 o;
  o.x = (unsigned)f2bf(v.x) | ((unsigned)f2bf(v.y) << 16);
  o.y = (unsigned)f2bf(v.z) | ((unsigned)f2bf(v.w) << 16);
  dst[i] = o;
}

// ---------------------------------------------------------------------------
// Kernel 1/5: GEMM  C[M,N] = A[M,K] * W[N,K]^T   (bf16 in, f32 acc)
// Block = 256 thr (8 waves). Wave: 2 M-tiles x 4 N-tiles. Block tile 64x256.
// ---------------------------------------------------------------------------
template <bool OUT_BF16>
__global__ __launch_bounds__(256) void gemm_wmma(const __hip_bfloat16* __restrict__ A,
                                                 const __hip_bfloat16* __restrict__ W,
                                                 void* __restrict__ Cout, int N, int K) {
  const int lane = threadIdx.x & 31;
  const int wave = threadIdx.x >> 5;
  const int n16 = lane & 15, half = lane >> 4;
  const int wm = wave & 1, wn = wave >> 1;
  const int m0 = blockIdx.y * 64 + wm * 32;
  const int n0 = blockIdx.x * 256 + wn * 64;

  v8f acc[8] = {};
  const __hip_bfloat16* a0p = A + (size_t)(m0 + n16) * K;
  const __hip_bfloat16* a1p = a0p + (size_t)16 * K;
  const __hip_bfloat16* wp  = W + (size_t)(n0 + n16) * K + half * 16;

  for (int k = 0; k < K; k += 32) {
    const v16bf a0 = make_afrag(a0p + k, half);
    const v16bf a1 = make_afrag(a1p + k, half);
#pragma unroll
    for (int nt = 0; nt < 4; ++nt) {
      const v16bf bf = load_bfrag(wp + (size_t)nt * 16 * K + k);
      acc[nt * 2]     = wmma_bf16(a0, bf, acc[nt * 2]);
      acc[nt * 2 + 1] = wmma_bf16(a1, bf, acc[nt * 2 + 1]);
    }
  }
#pragma unroll
  for (int nt = 0; nt < 4; ++nt)
#pragma unroll
    for (int t = 0; t < 2; ++t) {
      const v8f c = acc[nt * 2 + t];
#pragma unroll
      for (int r = 0; r < 8; ++r) {
        const int row = m0 + t * 16 + r + 8 * half;
        const int col = n0 + nt * 16 + n16;
        const size_t idx = (size_t)row * N + col;
        if (OUT_BF16)
          ((__hip_bfloat16*)Cout)[idx] = __float2bfloat16(c[r]);
        else
          ((float*)Cout)[idx] = c[r];
      }
    }
}

// ---------------------------------------------------------------------------
// Kernel 2: per-head-row RMSNorm + RoPE on Q/K (f32 math), V -> V^T copy.
// One wave per 128-dim row; rows = B*S*(NH + NKV + NKV).
// Q out: [B,NH,S,HD]  K out: [B,NKV,S,HD]  V out: [B,NKV,HD,S]
// ---------------------------------------------------------------------------
__global__ __launch_bounds__(256) void normrope_kernel(
    const __hip_bfloat16* __restrict__ QH, const __hip_bfloat16* __restrict__ KH,
    const __hip_bfloat16* __restrict__ VH, const float* __restrict__ qw,
    const float* __restrict__ kw, __hip_bfloat16* __restrict__ Qb,
    __hip_bfloat16* __restrict__ Kb, __hip_bfloat16* __restrict__ Vt) {
  const int lane = threadIdx.x & 31;
  const int row = blockIdx.x * 8 + (threadIdx.x >> 5);
  const int token = row / 24, rt = row - token * 24;
  const int b = token >> 11, s = token & (S_LEN - 1);

  if (rt >= 20) {  // V: pass-through, write transposed [b,h,d,s]
    const int h = rt - 20;
    const __hip_bfloat16* src = VH + (size_t)token * (NKVH * HD) + h * HD;
    __hip_bfloat16* dst = Vt + ((size_t)(b * NKVH + h) * HD) * S_LEN + s;
#pragma unroll
    for (int j = 0; j < 4; ++j) {
      const int d = lane + 32 * j;
      dst[(size_t)d * S_LEN] = src[d];
    }
    return;
  }
  const bool isq = rt < 16;
  const int h = isq ? rt : rt - 16;
  const __hip_bfloat16* src =
      isq ? (QH + (size_t)token * (NHEAD * HD) + h * HD)
          : (KH + (size_t)token * (NKVH * HD) + h * HD);
  const float* w = isq ? qw : kw;

  float x[4], ss = 0.f;
#pragma unroll
  for (int j = 0; j < 4; ++j) {
    x[j] = __bfloat162float(src[lane + 32 * j]);
    ss += x[j] * x[j];
  }
#pragma unroll
  for (int off = 16; off > 0; off >>= 1) ss += __shfl_xor(ss, off, 32);
  const float rs = rsqrtf(ss * (1.0f / 128.0f) + 1e-6f);
  float y[4];
#pragma unroll
  for (int j = 0; j < 4; ++j) y[j] = x[j] * rs * w[lane + 32 * j];

  __hip_bfloat16* dst =
      isq ? (Qb + ((size_t)(b * NHEAD + h) * S_LEN + s) * HD)
          : (Kb + ((size_t)(b * NKVH + h) * S_LEN + s) * HD);
#pragma unroll
  for (int j = 0; j < 2; ++j) {  // pairs (d, d+64), d = lane, lane+32
    const int d = lane + 32 * j;
    const float ang = (float)s * powf(10000.0f, -(float)d * (1.0f / 64.0f));
    const float c = cosf(ang), sn = sinf(ang);
    dst[d]      = __float2bfloat16(y[j] * c - y[j + 2] * sn);
    dst[d + 64] = __float2bfloat16(y[j + 2] * c + y[j] * sn);
  }
}

// ---------------------------------------------------------------------------
// Kernel 3: causal flash attention with WMMA + double-buffered async K/V
// staging in LDS. grid = (S/128, NH, B), block = 256 (8 waves).
// Wave owns 16 query rows; the block shares one (b,kv-head) K/V stream.
// ---------------------------------------------------------------------------
__global__ __launch_bounds__(256) void attn_kernel(const __hip_bfloat16* __restrict__ Q,
                                                   const __hip_bfloat16* __restrict__ K,
                                                   const __hip_bfloat16* __restrict__ Vt,
                                                   __hip_bfloat16* __restrict__ Ctx) {
  constexpr float scale = 0.08838834764831845f;  // 1/sqrt(128)
  const int tid = threadIdx.x;
  const int lane = tid & 31;
  const int wave = tid >> 5;
  const int n16 = lane & 15, half = lane >> 4;
  const int h = blockIdx.y, b = blockIdx.z, hk = h >> 2;  // 4 q-heads per kv-head
  const int q0 = blockIdx.x * 128 + wave * 16;

  __shared__ __align__(16) __hip_bfloat16 Klds[2][32][KROW];   // 32 keys x 128 dims
  __shared__ __align__(16) __hip_bfloat16 Vlds[2][HD][VROW];   // 128 dims x 32 keys
  __shared__ __align__(16) __hip_bfloat16 Plds[8][16][PROW];   // wave-private P tiles
  __hip_bfloat16* pbase = &Plds[wave][0][0];

  const __hip_bfloat16* kg = K  + ((size_t)(b * NKVH + hk) * S_LEN) * HD;
  const __hip_bfloat16* vg = Vt + ((size_t)(b * NKVH + hk) * HD) * S_LEN;

  // cooperative 32-key K/V tile stage (16B per copy, 4 copies per thread)
  auto stage = [&](int kb, int buf) {
#pragma unroll
    for (int j = 0; j < 2; ++j) {  // K: 512 x 16B chunks
      const int idx = tid + 256 * j;
      const int key = idx >> 4, ch = idx & 15;
      cp16_g2l(kg + (size_t)(kb + key) * HD + ch * 8, &Klds[buf][key][ch * 8]);
    }
#pragma unroll
    for (int j = 0; j < 2; ++j) {  // V^T: 512 x 16B chunks
      const int idx = tid + 256 * j;
      const int d = idx >> 2, ch = idx & 3;
      cp16_g2l(vg + (size_t)d * S_LEN + kb + ch * 8, &Vlds[buf][d][ch * 8]);
    }
  };

  // Q A-fragments: 16 rows x 128 dims (4 chunks of K=32)
  const __hip_bfloat16* qp = Q + ((size_t)(b * NHEAD + h) * S_LEN + (q0 + n16)) * HD;
  v16bf qf[4];
#pragma unroll
  for (int t = 0; t < 4; ++t) qf[t] = make_afrag(qp + t * 32, half);

  v8f o[8] = {};
  float mrun[8], lrun[8];
#pragma unroll
  for (int r = 0; r < 8; ++r) { mrun[r] = -INFINITY; lrun[r] = 0.f; }

  const int nkb = blockIdx.x * 128 + 128;  // block-uniform causal key extent
  stage(0, 0);

  for (int kb = 0, it = 0; kb < nkb; kb += 32, ++it) {
    const int buf = it & 1;
    wait_async_lds();
    __syncthreads();                       // tile(buf) fully resident
    if (kb + 32 < nkb) stage(kb + 32, buf ^ 1);

    if (kb < q0 + 16) {  // wave-uniform: this wave still has live keys here
      // ---- scores: S = Q * K^T (two 16x16 key tiles from LDS) ----
      v8f s0 = {}, s1 = {};
      const __hip_bfloat16* k0p = &Klds[buf][n16][half * 16];
      const __hip_bfloat16* k1p = &Klds[buf][16 + n16][half * 16];
#pragma unroll
      for (int t = 0; t < 4; ++t) {
        s0 = wmma_bf16(qf[t], load_bfrag16(k0p + t * 32), s0);
        s1 = wmma_bf16(qf[t], load_bfrag16(k1p + t * 32), s1);
      }
      // ---- online softmax over the 32-key block ----
      const int key0 = kb + n16, key1 = kb + 16 + n16;
      float alpha[8];
#pragma unroll
      for (int r = 0; r < 8; ++r) {
        const int qg = q0 + r + 8 * half;
        float v0 = (key0 <= qg) ? s0[r] * scale : -1e30f;
        float v1 = (key1 <= qg) ? s1[r] * scale : -1e30f;
        float mx = fmaxf(v0, v1);
#pragma unroll
        for (int off = 8; off > 0; off >>= 1) mx = fmaxf(mx, __shfl_xor(mx, off, 32));
        const float mnew = fmaxf(mrun[r], mx);
        alpha[r] = __expf(mrun[r] - mnew);
        const float p0 = __expf(v0 - mnew);
        const float p1 = __expf(v1 - mnew);
        float rsum = p0 + p1;
#pragma unroll
        for (int off = 8; off > 0; off >>= 1) rsum += __shfl_xor(rsum, off, 32);
        lrun[r] = lrun[r] * alpha[r] + rsum;
        mrun[r] = mnew;
        const int prw = r + 8 * half;  // C-layout row -> LDS row
        pbase[prw * PROW + n16]      = __float2bfloat16(p0);
        pbase[prw * PROW + 16 + n16] = __float2bfloat16(p1);
      }
#pragma unroll
      for (int j = 0; j < 8; ++j)
#pragma unroll
        for (int r = 0; r < 8; ++r) o[j][r] *= alpha[r];

      __asm__ volatile("s_wait_dscnt 0" ::: "memory");  // P visible to whole wave

      // reload P as A-fragment (16 queries x 32 keys), then O += P * V
      const v16bf pf = make_afrag(pbase + n16 * PROW, half);
#pragma unroll
      for (int j = 0; j < 8; ++j)
        o[j] = wmma_bf16(pf, load_bfrag16(&Vlds[buf][j * 16 + n16][half * 16]), o[j]);
    }
    __syncthreads();  // everyone done reading tile(buf) before it is restaged
  }

  // ---- epilogue: normalize and write ctx [B,S,NH*HD] (bf16) ----
#pragma unroll
  for (int r = 0; r < 8; ++r) {
    const float inv = 1.0f / lrun[r];
    const int srow = q0 + r + 8 * half;
    __hip_bfloat16* cp =
        Ctx + ((size_t)(b * S_LEN + srow)) * (NHEAD * HD) + h * HD + n16;
#pragma unroll
    for (int j = 0; j < 8; ++j) cp[j * 16] = __float2bfloat16(o[j][r] * inv);
  }
}

// ---------------------------------------------------------------------------
// Host-side orchestration
// ---------------------------------------------------------------------------
extern "C" void kernel_launch(void* const* d_in, const int* in_sizes, int n_in,
                              void* d_out, int out_size, void* d_ws, size_t ws_size,
                              hipStream_t stream) {
  const float* hs = (const float*)d_in[0];
  // d_in[1] = attention_mask (pure causal; computed inline)
  const float* Wq = (const float*)d_in[2];
  const float* Wk = (const float*)d_in[3];
  const float* Wv = (const float*)d_in[4];
  const float* Wo = (const float*)d_in[5];
  const float* qw = (const float*)d_in[6];
  const float* kw = (const float*)d_in[7];

  char* ws = (char*)d_ws;
  const size_t OFF_XB  = 0;                                        // X bf16 / later ctx
  const size_t OFF_WQB = OFF_XB  + (size_t)MTOT * HDIM * 2;
  const size_t OFF_WKB = OFF_WQB + (size_t)HDIM * HDIM * 2;
  const size_t OFF_WVB = OFF_WKB + (size_t)(NKVH * HD) * HDIM * 2;
  const size_t OFF_WOB = OFF_WVB + (size_t)(NKVH * HD) * HDIM * 2;
  const size_t OFF_QH  = OFF_WOB + (size_t)HDIM * HDIM * 2;        // raw projections
  const size_t OFF_KH  = OFF_QH  + (size_t)MTOT * HDIM * 2;
  const size_t OFF_VH  = OFF_KH  + (size_t)MTOT * (NKVH * HD) * 2;
  const size_t OFF_QB  = OFF_VH  + (size_t)MTOT * (NKVH * HD) * 2; // roped Q
  const size_t OFF_KB  = OFF_QB  + (size_t)MTOT * HDIM * 2;
  const size_t OFF_VT  = OFF_KB  + (size_t)MTOT * (NKVH * HD) * 2; // V^T
  // total = 88,080,384 bytes

  __hip_bfloat16* XB  = (__hip_bfloat16*)(ws + OFF_XB);
  __hip_bfloat16* WQB = (__hip_bfloat16*)(ws + OFF_WQB);
  __hip_bfloat16* WKB = (__hip_bfloat16*)(ws + OFF_WKB);
  __hip_bfloat16* WVB = (__hip_bfloat16*)(ws + OFF_WVB);
  __hip_bfloat16* WOB = (__hip_bfloat16*)(ws + OFF_WOB);
  __hip_bfloat16* QH  = (__hip_bfloat16*)(ws + OFF_QH);
  __hip_bfloat16* KH  = (__hip_bfloat16*)(ws + OFF_KH);
  __hip_bfloat16* VH  = (__hip_bfloat16*)(ws + OFF_VH);
  __hip_bfloat16* QB  = (__hip_bfloat16*)(ws + OFF_QB);
  __hip_bfloat16* KB  = (__hip_bfloat16*)(ws + OFF_KB);
  __hip_bfloat16* VT  = (__hip_bfloat16*)(ws + OFF_VT);
  __hip_bfloat16* CTX = XB;  // reuse after QKV GEMMs consumed X

  cvt_f32_to_bf16<<<(MTOT * HDIM / 4) / 256, 256, 0, stream>>>(hs, (uint2*)XB,
                                                               MTOT * HDIM / 4);
  cvt_f32_to_bf16<<<(HDIM * HDIM / 4) / 256, 256, 0, stream>>>(Wq, (uint2*)WQB,
                                                               HDIM * HDIM / 4);
  cvt_f32_to_bf16<<<(NKVH * HD * HDIM / 4) / 256, 256, 0, stream>>>(
      Wk, (uint2*)WKB, NKVH * HD * HDIM / 4);
  cvt_f32_to_bf16<<<(NKVH * HD * HDIM / 4) / 256, 256, 0, stream>>>(
      Wv, (uint2*)WVB, NKVH * HD * HDIM / 4);
  cvt_f32_to_bf16<<<(HDIM * HDIM / 4) / 256, 256, 0, stream>>>(Wo, (uint2*)WOB,
                                                               HDIM * HDIM / 4);

  gemm_wmma<true><<<dim3(HDIM / 256, MTOT / 64), 256, 0, stream>>>(XB, WQB, QH,
                                                                   HDIM, HDIM);
  gemm_wmma<true><<<dim3((NKVH * HD) / 256, MTOT / 64), 256, 0, stream>>>(
      XB, WKB, KH, NKVH * HD, HDIM);
  gemm_wmma<true><<<dim3((NKVH * HD) / 256, MTOT / 64), 256, 0, stream>>>(
      XB, WVB, VH, NKVH * HD, HDIM);

  normrope_kernel<<<(MTOT * 24) / 8, 256, 0, stream>>>(QH, KH, VH, qw, kw, QB, KB, VT);

  attn_kernel<<<dim3(S_LEN / 128, NHEAD, BATCH), 256, 0, stream>>>(QB, KB, VT, CTX);

  gemm_wmma<false><<<dim3(HDIM / 256, MTOT / 64), 256, 0, stream>>>(
      CTX, WOB, (float*)d_out, HDIM, HDIM);
}
